// XWTPhaseGNNV2Core_41867341201602
// MI455X (gfx1250) — matrix operand order repaired and checked by reading, not compiled, not run on hardware
//
#include <hip/hip_runtime.h>
#include <hip/hip_bf16.h>
#include <math.h>

// ---------------------------------------------------------------------------
// Problem constants (match reference)
// ---------------------------------------------------------------------------
constexpr int B_ = 32, C_ = 22, T_ = 512, F_ = 32, M_ = 3, H_ = 32, ENC_ = 16, NCLS_ = 4;
constexpr int E_  = C_ * (C_ - 1);   // 462
constexpr int BC_ = B_ * C_;         // 704
constexpr int FM_ = F_ * M_;         // 96
constexpr int CP_ = 32;              // channel rows padded to 2 WMMA M-tiles
constexpr float THETA_  = 0.7853981633974483f;
constexpr float BN_EPS_ = 1e-5f;

// edge kernel tiling: split F into FG_ groups per batch element
constexpr int FG_  = 4;
constexpr int FPG_ = F_ / FG_;       // 8 freqs per workgroup
constexpr int NWG_EDGE = B_ * FG_;   // 128 workgroups per step

// ---------------------------------------------------------------------------
// Workspace layout (bytes). Everything 256B-aligned by construction.
// ---------------------------------------------------------------------------
constexpr size_t OFF_STATE    = 0;                                   // BC_*H_ f32
constexpr size_t OFF_PREV     = OFF_STATE + (size_t)BC_ * H_ * 4;    // B_*H_ f32
constexpr size_t OFF_LAST     = OFF_PREV  + (size_t)B_ * H_ * 4;     // B_*H_ f32
constexpr size_t OFF_AGG      = OFF_LAST  + (size_t)B_ * H_ * 4;     // B_*C_*FM_ f32
constexpr size_t OFF_GATE     = OFF_AGG   + (size_t)B_ * C_ * FM_ * 4; // T_*NWG_EDGE f32
constexpr size_t OFF_PACK_S2F = OFF_GATE  + (size_t)T_ * NWG_EDGE * 4; // 6 tiles * 512 f16
constexpr size_t OFF_PACK_GIN = OFF_PACK_S2F + 6 * 512 * 2;            // 6 tiles * 512 f16
constexpr size_t OFF_PACK_WIH = OFF_PACK_GIN + 6 * 512 * 2;            // 6 tiles * 512 f16
constexpr size_t OFF_PACK_WHH = OFF_PACK_WIH + 6 * 512 * 2;            // 6 tiles * 512 f16
constexpr size_t OFF_H1       = OFF_PACK_WHH + 6 * 512 * 2;            // BC_*ENC_*T_ f32
constexpr size_t OFF_H2       = OFF_H1 + (size_t)BC_ * ENC_ * T_ * 4;  // BC_*ENC_*T_ f32
constexpr size_t OFF_LOCAL    = OFF_H2 + (size_t)BC_ * ENC_ * T_ * 4;  // T_*BC_*FM_ f32  (~138 MB)

// ---------------------------------------------------------------------------
// Optional CDNA5 async global->LDS DMA (ASYNCcnt-tracked). Guarded so the
// file compiles on toolchains without the builtins. Parameter type per the
// toolchain diagnostic: address-space-qualified int pointers.
// ---------------------------------------------------------------------------
#if defined(__HIP_DEVICE_COMPILE__) && \
    __has_builtin(__builtin_amdgcn_global_load_async_to_lds_b32) && \
    __has_builtin(__builtin_amdgcn_s_wait_asynccnt)
#define HAVE_ASYNC_LDS 1
typedef __attribute__((address_space(1))) int GInt;   // global
typedef __attribute__((address_space(3))) int LInt;   // LDS
#else
#define HAVE_ASYNC_LDS 0
#endif

// ---------------------------------------------------------------------------
// WMMA fragment types / helpers (CDNA5 v_wmma_f32_16x16x32_f16)
// ---------------------------------------------------------------------------
typedef __attribute__((ext_vector_type(16))) _Float16 v16h;
typedef __attribute__((ext_vector_type(8)))  float    v8f;

#define DEVI __device__ __forceinline__

DEVI v8f wmma_f16(v16h a, v16h b, v8f c) {
  // D = A(16x32,f16) * B(32x16,f16) + C(16x16,f32)
  return __builtin_amdgcn_wmma_f32_16x16x32_f16(
      /*neg_a=*/false, a, /*neg_b=*/false, b,
      /*c_mod=*/(short)0, c, /*reuse_a=*/false, /*reuse_b=*/false);
}

// Build an A fragment (16x32 f16, ISA 7.12.2 layout) from row-major f32 LDS.
// Caller guarantees rows [row0, row0+16) exist (tiles are zero-padded).
DEVI v16h a_frag(const float* S, int ld, int row0, int k0) {
  int lane = threadIdx.x & 31;
  int r  = row0 + (lane & 15);
  int hi = lane >> 4;
  v16h a;
#pragma unroll
  for (int v = 0; v < 8; ++v) {
    int kb = k0 + ((v < 4) ? 0 : 16) + ((v & 3) << 1) + (hi ? 8 : 0);
    a[2 * v]     = (_Float16)S[r * ld + kb];
    a[2 * v + 1] = (_Float16)S[r * ld + kb + 1];
  }
  return a;
}

DEVI v16h b_frag_load(const _Float16* pack, int tile) {
  int lane = threadIdx.x & 31;
  return ((const v16h*)pack)[tile * 32 + lane];
}

DEVI float sigmf(float x) { return 1.f / (1.f + __expf(-x)); }

// ---------------------------------------------------------------------------
// init: zero recurrent state and prev_sum
// ---------------------------------------------------------------------------
__global__ void k_init(float* state, float* prev) {
  int i = blockIdx.x * 256 + threadIdx.x;
  if (i < BC_ * H_) state[i] = 0.f;
  if (i < B_ * H_)  prev[i]  = 0.f;
}

// ---------------------------------------------------------------------------
// conv1 + bn1 + relu : (BC_,1,T) -> h1 (BC_,ENC_,T)
// ---------------------------------------------------------------------------
__global__ void k_conv1(const float* __restrict__ raw, const float* __restrict__ w,
                        const float* __restrict__ bc, const float* __restrict__ g,
                        const float* __restrict__ bt, const float* __restrict__ mn,
                        const float* __restrict__ vr, float* __restrict__ h1) {
  __shared__ float ws[ENC_ * 5], sc[ENC_], sh[ENC_];
  int tid = threadIdx.x;
  if (tid < ENC_ * 5) ws[tid] = w[tid];
  if (tid >= 128 && tid < 128 + ENC_) {
    int oc = tid - 128;
    float s = g[oc] * rsqrtf(vr[oc] + BN_EPS_);
    sc[oc] = s;
    sh[oc] = bt[oc] + (bc[oc] - mn[oc]) * s;   // fold conv bias into BN shift
  }
  __syncthreads();
  int gidx = blockIdx.x * 256 + tid;           // over BC_*T_
  int b = gidx / T_, t = gidx % T_;
  const float* x = raw + (size_t)b * T_;
  float xv[5];
#pragma unroll
  for (int k = 0; k < 5; ++k) {
    int ti = t + k - 2;
    xv[k] = (ti >= 0 && ti < T_) ? x[ti] : 0.f;
  }
#pragma unroll
  for (int oc = 0; oc < ENC_; ++oc) {
    float a = 0.f;
#pragma unroll
    for (int k = 0; k < 5; ++k) a += ws[oc * 5 + k] * xv[k];
    h1[((size_t)b * ENC_ + oc) * T_ + t] = fmaxf(a * sc[oc] + sh[oc], 0.f);
  }
}

// ---------------------------------------------------------------------------
// conv2 + bn2 + relu : h1 (BC_,ENC_,T) -> h2 (BC_,ENC_,T). LDS-tiled in T.
// ---------------------------------------------------------------------------
__global__ void k_conv2(const float* __restrict__ h1, const float* __restrict__ w,
                        const float* __restrict__ bc, const float* __restrict__ g,
                        const float* __restrict__ bt, const float* __restrict__ mn,
                        const float* __restrict__ vr, float* __restrict__ h2) {
  __shared__ float tile[ENC_][132];
  __shared__ float ws[ENC_ * ENC_ * 5], sc[ENC_], sh[ENC_];
  int b = blockIdx.x, t0 = blockIdx.y * 128, tid = threadIdx.x;
  for (int i = tid; i < ENC_ * ENC_ * 5; i += 128) ws[i] = w[i];
  if (tid < ENC_) {
    float s = g[tid] * rsqrtf(vr[tid] + BN_EPS_);
    sc[tid] = s;
    sh[tid] = bt[tid] + (bc[tid] - mn[tid]) * s;
  }
  for (int i = tid; i < ENC_ * 132; i += 128) {
    int ic = i / 132, tl = i % 132;
    int tt = t0 + tl - 2;
    tile[ic][tl] = (tt >= 0 && tt < T_) ? h1[((size_t)b * ENC_ + ic) * T_ + tt] : 0.f;
  }
  __syncthreads();
  int t = t0 + tid;
#pragma unroll 4
  for (int oc = 0; oc < ENC_; ++oc) {
    float a = 0.f;
#pragma unroll
    for (int ic = 0; ic < ENC_; ++ic)
#pragma unroll
      for (int k = 0; k < 5; ++k)
        a += ws[(oc * ENC_ + ic) * 5 + k] * tile[ic][tid + k];
    h2[((size_t)b * ENC_ + oc) * T_ + t] = fmaxf(a * sc[oc] + sh[oc], 0.f);
  }
}

// ---------------------------------------------------------------------------
// local_all[t][bc][j] = enc(bc,:,t) . loc_w[j,:] + loc_b[j]   (j = f*3+m)
// ---------------------------------------------------------------------------
__global__ void k_local(const float* __restrict__ h2, const float* __restrict__ lw_g,
                        const float* __restrict__ lb_g, float* __restrict__ local_all) {
  __shared__ float enc_t[ENC_][256];
  __shared__ float lw[FM_ * ENC_], lb[FM_];
  int b = blockIdx.x, t0 = blockIdx.y * 256, tid = threadIdx.x;
  for (int i = tid; i < FM_ * ENC_; i += 256) lw[i] = lw_g[i];
  if (tid < FM_) lb[tid] = lb_g[tid];
  for (int i = tid; i < ENC_ * 256; i += 256) {
    int e = i >> 8, tl = i & 255;
    enc_t[e][tl] = h2[((size_t)b * ENC_ + e) * T_ + t0 + tl];
  }
  __syncthreads();
  int t = t0 + tid;
  float ev[ENC_];
#pragma unroll
  for (int e = 0; e < ENC_; ++e) ev[e] = enc_t[e][tid];
  float* outp = local_all + ((size_t)t * BC_ + b) * FM_;
  for (int j = 0; j < FM_; ++j) {
    float a = lb[j];
#pragma unroll
    for (int e = 0; e < ENC_; ++e) a += lw[j * ENC_ + e] * ev[e];
    outp[j] = a;
  }
}

// ---------------------------------------------------------------------------
// pack: pre-swizzle f16 B fragments (ISA 7.12.2 B layout). For Y = X @ W^T,
// B[k][n] = W[n*K + k].  24 tiles x 32 lanes x 16 halves.
// ---------------------------------------------------------------------------
__global__ void k_pack(const float* __restrict__ s2f_w, const float* __restrict__ gin_w,
                       const float* __restrict__ wih,   const float* __restrict__ whh,
                       _Float16* pS, _Float16* pG, _Float16* pI, _Float16* pH) {
  int tid = blockIdx.x * 256 + threadIdx.x;
  if (tid >= 24 * 32) return;
  int tile = tid / 32, lane = tid % 32;
  int ln = lane & 15, hi = lane >> 4;
  const float* W; int K, kt, nt; _Float16* dst;
  if (tile < 6)       { W = s2f_w; K = 32; kt = 0;                 nt = tile;      dst = pS + tile * 512; }
  else if (tile < 12) { int l = tile - 6;  W = gin_w; K = 96; kt = l / 2; nt = l % 2; dst = pG + l * 512; }
  else if (tile < 18) { W = wih;   K = 32; kt = 0;                 nt = tile - 12; dst = pI + (tile - 12) * 512; }
  else                { W = whh;   K = 32; kt = 0;                 nt = tile - 18; dst = pH + (tile - 18) * 512; }
#pragma unroll
  for (int e = 0; e < 16; ++e) {
    int v = e >> 1;
    int k = kt * 32 + ((v < 4) ? 0 : 16) + (e & 7) + (hi ? 8 : 0);
    int n = nt * 16 + ln;
    dst[lane * 16 + e] = (_Float16)W[n * K + k];
  }
}

// ---------------------------------------------------------------------------
// edge kernel (per step): one WG per (b, f-group of 8).
//   phase2: freq_state[b] = state[b] @ s2f^T + s2f_b   via 12 WMMAs
//   phase3: per (dst,f) thread, 21-src edge loop -> agg, gate partial
// ---------------------------------------------------------------------------
__global__ void k_edge(const float* __restrict__ wr, const float* __restrict__ wi,
                       const float* __restrict__ raw, const float* __restrict__ state_g,
                       const _Float16* __restrict__ pS, const float* __restrict__ s2fb_g,
                       const float* __restrict__ w1_g, const float* __restrict__ b1_g,
                       const float* __restrict__ w2_g, const float* __restrict__ b2_g,
                       float* __restrict__ agg_g, float* __restrict__ gate_part, int t) {
  __shared__ float st[CP_ * H_];         // old state, padded to 32 rows
  __shared__ float fs[CP_ * FM_];        // freq_state, padded to 32 rows
  __shared__ float wrl[C_ * FPG_], wil[C_ * FPG_], rt[C_], s2fb[FM_];
  __shared__ float w1s[M_ * 9], b1s[M_], w2s[M_ * M_], b2s[M_];
  __shared__ float gred[256];

  int b  = blockIdx.x / FG_;
  int f0 = (blockIdx.x % FG_) * FPG_;
  int tid = threadIdx.x;

  // ---- phase 1: stage (w slices via async DMA to LDS when available) ----
  for (int i = tid; i < CP_ * H_; i += 256)
    st[i] = (i < C_ * H_) ? state_g[b * C_ * H_ + i] : 0.f;
#if HAVE_ASYNC_LDS
  for (int i = tid; i < C_ * FPG_; i += 256) {
    int ch = i / FPG_, fi = i % FPG_;
    size_t gidx = (((size_t)(b * C_ + ch) * T_ + t) * F_) + f0 + fi;
    __builtin_amdgcn_global_load_async_to_lds_b32((GInt*)(wr + gidx), (LInt*)&wrl[i], 0, 0);
    __builtin_amdgcn_global_load_async_to_lds_b32((GInt*)(wi + gidx), (LInt*)&wil[i], 0, 0);
  }
  __builtin_amdgcn_s_wait_asynccnt(0);
#else
  for (int i = tid; i < C_ * FPG_; i += 256) {
    int ch = i / FPG_, fi = i % FPG_;
    size_t gidx = (((size_t)(b * C_ + ch) * T_ + t) * F_) + f0 + fi;
    wrl[i] = wr[gidx];
    wil[i] = wi[gidx];
  }
#endif
  if (tid < C_) rt[tid] = raw[(size_t)(b * C_ + tid) * T_ + t];
  for (int i = tid; i < FM_; i += 256) s2fb[i] = s2fb_g[i];
  if (tid < M_ * 9) w1s[tid] = w1_g[tid];
  if (tid < M_ * M_) w2s[tid] = w2_g[tid];
  if (tid < M_) { b1s[tid] = b1_g[tid]; b2s[tid] = b2_g[tid]; }
  __syncthreads();

  // ---- phase 2: freq_state via WMMA (2 mtiles x 6 ntiles over 8 waves) ----
  int wave = tid >> 5, lane = tid & 31;
  int ln = lane & 15, hi = lane >> 4;
  for (int tile = wave; tile < 12; tile += 8) {
    int mt = tile / 6, nt = tile % 6;
    v16h a  = a_frag(st, H_, mt * 16, 0);
    v16h bb = b_frag_load(pS, nt);
    v8f c = {};
    c = wmma_f16(a, bb, c);
    int n = nt * 16 + ln;
    float sb = s2fb[n];
    int m0 = mt * 16 + (hi ? 8 : 0);
#pragma unroll
    for (int v = 0; v < 8; ++v) fs[(m0 + v) * FM_ + n] = c[v] + sb;  // padded: no guard
  }
  __syncthreads();

  // ---- phase 3: edge loop grouped by destination ----
  float gsum = 0.f;
  if (tid < C_ * FPG_) {
    int dst = tid / FPG_, fi = tid % FPG_;
    int f = f0 + fi;
    float dr = wrl[dst * FPG_ + fi], di = wil[dst * FPG_ + fi];
    float rtd = rt[dst];
    const float* fsd = &fs[dst * FM_ + f * M_];
    float fd0 = fsd[0], fd1 = fsd[1], fd2 = fsd[2];
    float a0 = 0.f, a1 = 0.f, a2 = 0.f;
    for (int src = 0; src < C_; ++src) {
      if (src == dst) continue;
      float sr = wrl[src * FPG_ + fi], si = wil[src * FPG_ + fi];
      float xr = sr * dr + si * di;
      float xi = si * dr - sr * di;
      float mag = log1pf(sqrtf(xr * xr + xi * xi + 1e-12f));
      // gate: atan2(xi,xr) > pi/4 without the atan2 (angle region test):
      //   angle in (pi/4, pi]  <=>  (xi>0 && (xr<=0 || xi>xr)) || (xi==0 && xr<0)
      bool gb = (xi > 0.f) ? ((xr <= 0.f) || (xi > xr)) : ((xi == 0.f) && (xr < 0.f));
      float gate = gb ? 1.f : 0.f;
      gsum += gate;
      const float* fss = &fs[src * FM_ + f * M_];
      float ft[9] = {mag, fss[0], fss[1], fss[2], fd0, fd1, fd2, rt[src], rtd};
      float hm[3];
#pragma unroll
      for (int j = 0; j < 3; ++j) {
        float acc = b1s[j];
#pragma unroll
        for (int i = 0; i < 9; ++i) acc += w1s[j * 9 + i] * ft[i];
        hm[j] = fmaxf(acc, 0.f);
      }
      a0 += (b2s[0] + w2s[0] * hm[0] + w2s[1] * hm[1] + w2s[2] * hm[2]) * gate;
      a1 += (b2s[1] + w2s[3] * hm[0] + w2s[4] * hm[1] + w2s[5] * hm[2]) * gate;
      a2 += (b2s[2] + w2s[6] * hm[0] + w2s[7] * hm[1] + w2s[8] * hm[2]) * gate;
    }
    size_t o = ((size_t)(b * C_ + dst) * F_ + f) * M_;
    agg_g[o + 0] = a0; agg_g[o + 1] = a1; agg_g[o + 2] = a2;
  }
  gred[tid] = gsum;
  __syncthreads();
  for (int s = 128; s > 0; s >>= 1) {
    if (tid < s) gred[tid] += gred[tid + s];
    __syncthreads();
  }
  if (tid == 0) gate_part[t * NWG_EDGE + blockIdx.x] = gred[0];
}

// ---------------------------------------------------------------------------
// update kernel (per step): one WG per b (64 threads = 2 waves = 2 M-tiles).
//   upd = (agg+local) @ gin^T + gin_b        (6 WMMAs / wave)
//   gi  = upd @ wih^T ; gh = h0 @ whh^T      (12 WMMAs / wave)
//   GRU elementwise fused on C-fragments; pooled mean over channels.
// ---------------------------------------------------------------------------
__global__ void k_update(const float* __restrict__ agg_g, const float* __restrict__ local_all,
                         const _Float16* __restrict__ pG, const _Float16* __restrict__ pI,
                         const _Float16* __restrict__ pH, const float* __restrict__ ginb_g,
                         const float* __restrict__ bih_g, const float* __restrict__ bhh_g,
                         float* __restrict__ state_g, float* __restrict__ prev_sum,
                         float* __restrict__ last_pooled, int t) {
  __shared__ float X[CP_ * FM_];    // agg + local, padded to 32 rows
  __shared__ float st[CP_ * H_];    // old state, padded
  __shared__ float U[2 * 16 * H_];  // upd per mtile (rows padded to 16)
  __shared__ float ns[CP_ * H_];    // new state, padded
  __shared__ float gb[H_], bi[FM_], bh[FM_];

  int b = blockIdx.x, tid = threadIdx.x;
  const float* loc = local_all + ((size_t)t * BC_ + b * C_) * FM_;
  const float* ag  = agg_g + (size_t)b * C_ * FM_;
  for (int i = tid; i < CP_ * FM_; i += 64)
    X[i] = (i < C_ * FM_) ? (ag[i] + loc[i]) : 0.f;
  for (int i = tid; i < CP_ * H_; i += 64)
    st[i] = (i < C_ * H_) ? state_g[b * C_ * H_ + i] : 0.f;
  for (int i = tid; i < FM_; i += 64) { bi[i] = bih_g[i]; bh[i] = bhh_g[i]; }
  if (tid < H_) gb[tid] = ginb_g[tid];
  if (t + 1 < T_)   // warm L2 for next step's local slab (global_prefetch_b8)
    __builtin_prefetch(local_all + ((size_t)(t + 1) * BC_ + b * C_) * FM_ + tid * 32, 0, 1);
  __syncthreads();

  int wave = tid >> 5, lane = tid & 31;
  int ln = lane & 15, hi = lane >> 4;
  int mt = wave;                      // mtile 0: rows 0..15, mtile 1: rows 16..21(+pad)

  // ---- gin: X(22x96) @ gin^T -> upd(22x32) ----
  v16h a0 = a_frag(X, FM_, mt * 16, 0);
  v16h a1 = a_frag(X, FM_, mt * 16, 32);
  v16h a2 = a_frag(X, FM_, mt * 16, 64);
  float* Uw = U + mt * 16 * H_;
#pragma unroll
  for (int nt = 0; nt < 2; ++nt) {
    v8f c = {};
    c = wmma_f16(a0, b_frag_load(pG, 0 * 2 + nt), c);
    c = wmma_f16(a1, b_frag_load(pG, 1 * 2 + nt), c);
    c = wmma_f16(a2, b_frag_load(pG, 2 * 2 + nt), c);
    int n = nt * 16 + ln;
    float gbn = gb[n];
    int r0 = (hi ? 8 : 0);
#pragma unroll
    for (int v = 0; v < 8; ++v) Uw[(r0 + v) * H_ + n] = c[v] + gbn;
  }
  __syncthreads();

  // ---- GRU GEMMs + fused gates ----
  v16h au = a_frag(Uw, H_, 0, 0);
  v16h as = a_frag(st, H_, mt * 16, 0);
#pragma unroll
  for (int nt = 0; nt < 2; ++nt) {
    v8f gir = {}, ghr = {}, giz = {}, ghz = {}, gin_ = {}, ghn = {};
    gir  = wmma_f16(au, b_frag_load(pI, nt),     gir);
    ghr  = wmma_f16(as, b_frag_load(pH, nt),     ghr);
    giz  = wmma_f16(au, b_frag_load(pI, nt + 2), giz);
    ghz  = wmma_f16(as, b_frag_load(pH, nt + 2), ghz);
    gin_ = wmma_f16(au, b_frag_load(pI, nt + 4), gin_);
    ghn  = wmma_f16(as, b_frag_load(pH, nt + 4), ghn);
    int n = nt * 16 + ln;
    // hoist biases and h0 reads before the ns stores (avoid LDS reload churn)
    float bir = bi[n],      bhr = bh[n];
    float biz = bi[n + 32], bhz = bh[n + 32];
    float bin = bi[n + 64], bhn = bh[n + 64];
    int r0 = mt * 16 + (hi ? 8 : 0);
    float h0v[8];
#pragma unroll
    for (int v = 0; v < 8; ++v) h0v[v] = st[(r0 + v) * H_ + n];
#pragma unroll
    for (int v = 0; v < 8; ++v) {
      float r  = sigmf(gir[v] + bir + ghr[v] + bhr);
      float z  = sigmf(giz[v] + biz + ghz[v] + bhz);
      float nn = tanhf(gin_[v] + bin + r * (ghn[v] + bhn));
      ns[(r0 + v) * H_ + n] = (1.f - z) * nn + z * h0v[v];   // padded: no guard
    }
  }
  __syncthreads();

  // ---- coalesced state writeback + channel pooling + running sums ----
  for (int i = tid; i < C_ * H_; i += 64) state_g[b * C_ * H_ + i] = ns[i];
  if (tid < H_) {
    float s = 0.f;
    for (int c = 0; c < C_; ++c) s += ns[c * H_ + tid];
    float pooled = s / (float)C_;
    last_pooled[b * H_ + tid] = pooled;
    prev_sum[b * H_ + tid]   += pooled;
  }
}

// ---------------------------------------------------------------------------
// final: logits + gate_rate (deterministic tree reduce of per-WG partials)
// ---------------------------------------------------------------------------
__global__ void k_final(const float* __restrict__ last_pooled, const float* __restrict__ prev_sum,
                        const float* __restrict__ cls_w, const float* __restrict__ cls_b,
                        const float* __restrict__ gate_part, float* __restrict__ out) {
  __shared__ float red[256];
  int tid = threadIdx.x;
  float s = 0.f;
  for (int i = tid; i < T_ * NWG_EDGE; i += 256) s += gate_part[i];
  red[tid] = s;
  __syncthreads();
  for (int st = 128; st > 0; st >>= 1) {
    if (tid < st) red[tid] += red[tid + st];
    __syncthreads();
  }
  if (tid == 0)
    out[B_ * NCLS_] = red[0] / ((float)T_ * (float)B_ * (float)E_ * (float)F_);
  if (tid < B_ * NCLS_) {
    int b = tid / NCLS_, n = tid % NCLS_;
    float a = cls_b[n];
    for (int k = 0; k < H_; ++k) a += cls_w[n * 2 * H_ + k] * last_pooled[b * H_ + k];
    for (int k = 0; k < H_; ++k) a += cls_w[n * 2 * H_ + H_ + k] * (prev_sum[b * H_ + k] / (float)T_);
    out[tid] = a;
  }
}

// ---------------------------------------------------------------------------
// host launcher
// ---------------------------------------------------------------------------
extern "C" void kernel_launch(void* const* d_in, const int* in_sizes, int n_in,
                              void* d_out, int out_size, void* d_ws, size_t ws_size,
                              hipStream_t stream) {
  // inputs: raw_x, w_real, w_imag, then params in dict order
  const float* raw    = (const float*)d_in[0];
  const float* wr     = (const float*)d_in[1];
  const float* wi     = (const float*)d_in[2];
  const float* conv1w = (const float*)d_in[3];
  const float* conv1b = (const float*)d_in[4];
  const float* bn1g   = (const float*)d_in[5];
  const float* bn1b   = (const float*)d_in[6];
  const float* bn1m   = (const float*)d_in[7];
  const float* bn1v   = (const float*)d_in[8];
  const float* conv2w = (const float*)d_in[9];
  const float* conv2b = (const float*)d_in[10];
  const float* bn2g   = (const float*)d_in[11];
  const float* bn2b   = (const float*)d_in[12];
  const float* bn2m   = (const float*)d_in[13];
  const float* bn2v   = (const float*)d_in[14];
  const float* msgw1  = (const float*)d_in[15];
  const float* msgb1  = (const float*)d_in[16];
  const float* msgw2  = (const float*)d_in[17];
  const float* msgb2  = (const float*)d_in[18];
  const float* locw   = (const float*)d_in[19];
  const float* locb   = (const float*)d_in[20];
  const float* s2fw   = (const float*)d_in[21];
  const float* s2fb   = (const float*)d_in[22];
  const float* ginw   = (const float*)d_in[23];
  const float* ginb   = (const float*)d_in[24];
  const float* wih    = (const float*)d_in[25];
  const float* whh    = (const float*)d_in[26];
  const float* bih    = (const float*)d_in[27];
  const float* bhh    = (const float*)d_in[28];
  const float* clsw   = (const float*)d_in[29];
  const float* clsb   = (const float*)d_in[30];

  char* ws = (char*)d_ws;
  float*    state  = (float*)(ws + OFF_STATE);
  float*    prev   = (float*)(ws + OFF_PREV);
  float*    lastp  = (float*)(ws + OFF_LAST);
  float*    agg    = (float*)(ws + OFF_AGG);
  float*    gatep  = (float*)(ws + OFF_GATE);
  _Float16* pS     = (_Float16*)(ws + OFF_PACK_S2F);
  _Float16* pG     = (_Float16*)(ws + OFF_PACK_GIN);
  _Float16* pI     = (_Float16*)(ws + OFF_PACK_WIH);
  _Float16* pH     = (_Float16*)(ws + OFF_PACK_WHH);
  float*    h1     = (float*)(ws + OFF_H1);
  float*    h2     = (float*)(ws + OFF_H2);
  float*    localA = (float*)(ws + OFF_LOCAL);
  float*    out    = (float*)d_out;

  k_init<<<(BC_ * H_ + 255) / 256, 256, 0, stream>>>(state, prev);
  k_conv1<<<(BC_ * T_) / 256, 256, 0, stream>>>(raw, conv1w, conv1b, bn1g, bn1b, bn1m, bn1v, h1);
  k_conv2<<<dim3(BC_, T_ / 128), 128, 0, stream>>>(h1, conv2w, conv2b, bn2g, bn2b, bn2m, bn2v, h2);
  k_local<<<dim3(BC_, T_ / 256), 256, 0, stream>>>(h2, locw, locb, localA);
  k_pack<<<3, 256, 0, stream>>>(s2fw, ginw, wih, whh, pS, pG, pI, pH);

  for (int t = 0; t < T_; ++t) {
    k_edge<<<NWG_EDGE, 256, 0, stream>>>(wr, wi, raw, state, pS, s2fb,
                                         msgw1, msgb1, msgw2, msgb2, agg, gatep, t);
    k_update<<<B_, 64, 0, stream>>>(agg, localA, pG, pI, pH, ginb, bih, bhh,
                                    state, prev, lastp, t);
  }
  k_final<<<1, 256, 0, stream>>>(lastp, prev, clsw, clsb, gatep, out);
}
